// _StackedGIN_55568286876150
// MI455X (gfx1250) — matrix operand order repaired and checked by reading, not compiled, not run on hardware
//
#include <hip/hip_runtime.h>
#include <hip/hip_bf16.h>

// ---------------------------------------------------------------------------
// StackedGIN forward for MI455X (gfx1250, wave32, WMMA).
//
// Cost model: GEMMs are only 5.6 GFLOP -> V_WMMA_F32_16X16X32_BF16 (bf16 in,
// fp32 accumulate). The real cost is edge aggregation: 307 MB/layer of
// gathered rows + 76.8M float atomics, all L2-resident (h = 19.2 MB << 192 MB
// L2), done with float4 gathers + native no-return global_atomic_add_f32
// (fp32 atomics keep sum-aggregation exact-ish; bf16 pk atomics would lose
// ~2^-9 * degree relative accuracy). Weights are pre-shuffled on device into
// WMMA B-fragment register order so each wave loads its fragment as one
// contiguous 32 B per lane (global_load_b128 pairs).
// ---------------------------------------------------------------------------

typedef __attribute__((ext_vector_type(16))) __bf16 v16bf;
typedef __attribute__((ext_vector_type(8)))  float  v8f;

#define GIN_D 96   // feature dim (in == hidden)
#define GIN_C 10   // classes
#define GIN_L 3    // layers

// Fragment geometry: 3 k-steps x 6 n-tiles x 32 lanes x 16 bf16 = 9216 = D*D
#define FRAG_PER_LAYER (3 * 6 * 32 * 16)

// ---------------------------------------------------------------------------
// Pack fp32 weights (L,D,D) row-major [k][n] into bf16 WMMA B-fragment order:
//   flat = (((l*3 + ks)*6 + nt)*32 + lane)*16 + idx
// where per CDNA5 ISA 7.12.2 (B 32x16 bf16, wave32):
//   n16 = lane&15, khalf = lane>>4, v = idx>>1, odd = idx&1
//   k = ks*32 + (v<4?0:16) + khalf*8 + (v&3)*2 + odd,  col = nt*16 + n16
// ---------------------------------------------------------------------------
__global__ __launch_bounds__(256)
void gin_pack_wfrag(const float* __restrict__ W, __bf16* __restrict__ out,
                    int total) {
  int i = blockIdx.x * blockDim.x + threadIdx.x;
  if (i >= total) return;
  const int idx  = i & 15;
  const int lane = (i >> 4) & 31;
  const int nt   = (i >> 9) % 6;
  const int rest = (i >> 9) / 6;     // = l*3 + ks
  const int ks   = rest % 3;
  const int l    = rest / 3;
  const int n16 = lane & 15, khalf = lane >> 4;
  const int v = idx >> 1, odd = idx & 1;
  const int k   = ks * 32 + (v < 4 ? 0 : 16) + khalf * 8 + (v & 3) * 2 + odd;
  const int col = nt * 16 + n16;
  out[i] = (__bf16)W[((size_t)l * GIN_D + k) * GIN_D + col];
}

// ---------------------------------------------------------------------------
// Edge scatter: agg[dst] += h[src]  (agg pre-initialized to h -> (1+eps)*h,
// eps=0). One thread per (edge, 4-feature chunk): 96/4 = 24 chunks.
// ---------------------------------------------------------------------------
__global__ __launch_bounds__(256)
void gin_scatter(const float* __restrict__ h, float* __restrict__ agg,
                 const int* __restrict__ src, const int* __restrict__ dst,
                 int nedges) {
  long long i = (long long)blockIdx.x * blockDim.x + threadIdx.x;
  long long items = (long long)nedges * 24;
  if (i >= items) return;
  int e = (int)(i / 24);
  int q = (int)(i % 24);
  int s = src[e];
  int d = dst[e];
  const float4 v = *(const float4*)(h + (size_t)s * GIN_D + q * 4);
  float* a = agg + (size_t)d * GIN_D + q * 4;
  __hip_atomic_fetch_add(a + 0, v.x, __ATOMIC_RELAXED, __HIP_MEMORY_SCOPE_AGENT);
  __hip_atomic_fetch_add(a + 1, v.y, __ATOMIC_RELAXED, __HIP_MEMORY_SCOPE_AGENT);
  __hip_atomic_fetch_add(a + 2, v.z, __ATOMIC_RELAXED, __HIP_MEMORY_SCOPE_AGENT);
  __hip_atomic_fetch_add(a + 3, v.w, __ATOMIC_RELAXED, __HIP_MEMORY_SCOPE_AGENT);
}

// ---------------------------------------------------------------------------
// Fused per-layer MLP: hout = relu( relu(agg@W1 + B1) @ W2 + B2 )
// Block = 192 threads = 6 full waves (EXEC all-ones, as WMMA requires);
// block owns a 16-node tile; each wave owns one 16-wide output slice.
// K=96 -> 3 x wmma_f32_16x16x32_bf16 per GEMM.
//
// Fragment layouts per CDNA5 ISA 7.12.2 (wave32):
//   A(16x32 bf16): lane l: m=l&15, half=l>>4;
//                  vgpr v packs K = (v<4?0:16) + half*8 + (v&3)*2, +1
//                  -> two contiguous 16 B chunks per lane (ds_load_b128 x2)
//   B(32x16 bf16): pre-shuffled fragments, 32 B contiguous per lane
//   C/D(16x16 f32): lane l: n=l&15; vgpr r holds m = r + 8*(l>>4)
// ---------------------------------------------------------------------------
__global__ __launch_bounds__(192)
void gin_mlp(const float* __restrict__ agg,
             const __bf16* __restrict__ W1f, const float* __restrict__ B1,
             const __bf16* __restrict__ W2f, const float* __restrict__ B2,
             float* __restrict__ hout) {
  __shared__ __bf16 ldsA[16 * GIN_D];
  __shared__ __bf16 ldsZ[16 * GIN_D];

  const int tid  = threadIdx.x;
  const int row0 = blockIdx.x * 16;

  // Stage agg tile (16 x 96 fp32 -> bf16 LDS), 8 contiguous floats per
  // thread: 2x global_load_b128 -> 4x v_cvt_pk_bf16_f32 -> one 16 B LDS store.
  {
    const int r  = tid / 12;          // row within tile
    const int cq = tid % 12;          // 8-float chunk within row
    const float4* g =
        (const float4*)(agg + (size_t)(row0 + r) * GIN_D + cq * 8);
    const float4 lo = g[0];
    const float4 hi = g[1];
    union { __bf16 b[8]; uint4 u; } pk;
    pk.b[0] = (__bf16)lo.x; pk.b[1] = (__bf16)lo.y;
    pk.b[2] = (__bf16)lo.z; pk.b[3] = (__bf16)lo.w;
    pk.b[4] = (__bf16)hi.x; pk.b[5] = (__bf16)hi.y;
    pk.b[6] = (__bf16)hi.z; pk.b[7] = (__bf16)hi.w;
    *(uint4*)(ldsA + r * GIN_D + cq * 8) = pk.u;
  }
  __syncthreads();

  const int lane = tid & 31;
  const int wv   = tid >> 5;        // 0..5 : output column tile
  const int half = lane >> 4;
  const int n16  = lane & 15;
  const int m16  = lane & 15;
  const int ncol = wv * 16 + n16;

  const v16bf* W1v = (const v16bf*)W1f;
  const v16bf* W2v = (const v16bf*)W2f;

  // ---------------- GEMM 1: Z = relu(A @ W1 + B1) ----------------
  v8f acc;
  {
    const float b = B1[ncol];
#pragma unroll
    for (int r = 0; r < 8; ++r) acc[r] = b;
  }
#pragma unroll
  for (int ks = 0; ks < 3; ++ks) {
    v16bf af;
#pragma unroll
    for (int v = 0; v < 8; ++v) {
      const int kb = ks * 32 + (v < 4 ? 0 : 16) + half * 8 + (v & 3) * 2;
      af[2 * v]     = ldsA[m16 * GIN_D + kb];
      af[2 * v + 1] = ldsA[m16 * GIN_D + kb + 1];
    }
    const v16bf bfr = W1v[(ks * 6 + wv) * 32 + lane];
    acc = __builtin_amdgcn_wmma_f32_16x16x32_bf16(
        false, af, false, bfr, (short)0, acc, false, false);
  }
  // ReLU -> Z tile (bf16) in LDS
#pragma unroll
  for (int r = 0; r < 8; ++r) {
    const int m = r + 8 * half;
    ldsZ[m * GIN_D + ncol] = (__bf16)__builtin_fmaxf(acc[r], 0.0f);
  }
  __syncthreads();

  // ---------------- GEMM 2: H = relu(Z @ W2 + B2) ----------------
  {
    const float b = B2[ncol];
#pragma unroll
    for (int r = 0; r < 8; ++r) acc[r] = b;
  }
#pragma unroll
  for (int ks = 0; ks < 3; ++ks) {
    v16bf af;
#pragma unroll
    for (int v = 0; v < 8; ++v) {
      const int kb = ks * 32 + (v < 4 ? 0 : 16) + half * 8 + (v & 3) * 2;
      af[2 * v]     = ldsZ[m16 * GIN_D + kb];
      af[2 * v + 1] = ldsZ[m16 * GIN_D + kb + 1];
    }
    const v16bf bfr = W2v[(ks * 6 + wv) * 32 + lane];
    acc = __builtin_amdgcn_wmma_f32_16x16x32_bf16(
        false, af, false, bfr, (short)0, acc, false, false);
  }
  // ReLU -> global h
#pragma unroll
  for (int r = 0; r < 8; ++r) {
    const int m = r + 8 * half;
    hout[(size_t)(row0 + m) * GIN_D + ncol] = __builtin_fmaxf(acc[r], 0.0f);
  }
}

// ---------------------------------------------------------------------------
// Classifier: out = h @ Wc + Bc   (50000 x 96 @ 96 x 10 -> 96 MFLOP, trivial)
// ---------------------------------------------------------------------------
__global__ __launch_bounds__(256)
void gin_classifier(const float* __restrict__ h, const float* __restrict__ Wc,
                    const float* __restrict__ Bc, float* __restrict__ out,
                    int nnodes) {
  int i = blockIdx.x * blockDim.x + threadIdx.x;
  if (i >= nnodes * GIN_C) return;
  const int node = i / GIN_C;
  const int c    = i % GIN_C;
  float a = Bc[c];
  const float* hr = h + (size_t)node * GIN_D;
#pragma unroll 8
  for (int k = 0; k < GIN_D; ++k) a = __builtin_fmaf(hr[k], Wc[k * GIN_C + c], a);
  out[i] = a;
}

// ---------------------------------------------------------------------------
static inline size_t alignup(size_t x) { return (x + 255) & ~(size_t)255; }

extern "C" void kernel_launch(void* const* d_in, const int* in_sizes, int n_in,
                              void* d_out, int out_size, void* d_ws, size_t ws_size,
                              hipStream_t stream) {
  const float* x  = (const float*)d_in[0];
  const int*   ei = (const int*)d_in[1];   // [2,E]: src = ei[0:E], dst = ei[E:2E]
  const float* W1 = (const float*)d_in[2]; // (L,D,D)
  const float* B1 = (const float*)d_in[3]; // (L,D)
  const float* W2 = (const float*)d_in[4]; // (L,D,D)
  const float* B2 = (const float*)d_in[5]; // (L,D)
  const float* Wc = (const float*)d_in[6]; // (D,C)
  const float* Bc = (const float*)d_in[7]; // (C,)
  float* out = (float*)d_out;

  const int nnodes = in_sizes[0] / GIN_D;  // 50000
  const int nedges = in_sizes[1] / 2;      // 800000

  // Workspace layout
  char* ws = (char*)d_ws;
  size_t off = 0;
  float* agg = (float*)(ws + off); off += alignup((size_t)nnodes * GIN_D * sizeof(float));
  float* h   = (float*)(ws + off); off += alignup((size_t)nnodes * GIN_D * sizeof(float));
  __bf16* W1b = (__bf16*)(ws + off); off += alignup((size_t)GIN_L * FRAG_PER_LAYER * sizeof(__bf16));
  __bf16* W2b = (__bf16*)(ws + off); off += alignup((size_t)GIN_L * FRAG_PER_LAYER * sizeof(__bf16));

  // 1) Pack MLP weights into bf16 WMMA fragments (deterministic, per call).
  {
    const int nw = GIN_L * FRAG_PER_LAYER;   // 27648
    gin_pack_wfrag<<<(nw + 255) / 256, 256, 0, stream>>>(W1, W1b, nw);
    gin_pack_wfrag<<<(nw + 255) / 256, 256, 0, stream>>>(W2, W2b, nw);
  }

  // 2) h = x
  hipMemcpyAsync(h, x, (size_t)nnodes * GIN_D * sizeof(float),
                 hipMemcpyDeviceToDevice, stream);

  // 3) Layers
  const long long scatter_items = (long long)nedges * 24;
  const int scatter_blocks = (int)((scatter_items + 255) / 256);
  const int mtiles = nnodes / 16;  // 50000 % 16 == 0 -> 3125
  for (int l = 0; l < GIN_L; ++l) {
    // agg = (1+eps)*h, eps=0
    hipMemcpyAsync(agg, h, (size_t)nnodes * GIN_D * sizeof(float),
                   hipMemcpyDeviceToDevice, stream);
    gin_scatter<<<scatter_blocks, 256, 0, stream>>>(h, agg, ei, ei + nedges, nedges);
    gin_mlp<<<mtiles, 192, 0, stream>>>(
        agg,
        W1b + (size_t)l * FRAG_PER_LAYER, B1 + (size_t)l * GIN_D,
        W2b + (size_t)l * FRAG_PER_LAYER, B2 + (size_t)l * GIN_D,
        h);
  }

  // 4) Classifier
  gin_classifier<<<(nnodes * GIN_C + 255) / 256, 256, 0, stream>>>(h, Wc, Bc, out, nnodes);
}
